// RNN_81260781240589
// MI455X (gfx1250) — matrix operand order repaired and checked by reading, not compiled, not run on hardware
//
#include <hip/hip_runtime.h>
#include <hip/hip_bf16.h>

// Dims (hardcoded from reference): X_DIM=512, H=1024, Y=512, B=64, T=512.
typedef __attribute__((ext_vector_type(16))) _Float16 v16h;
typedef __attribute__((ext_vector_type(8)))  _Float16 v8h;
typedef __attribute__((ext_vector_type(8)))  float    v8f;
typedef unsigned int uint32x4 __attribute__((ext_vector_type(4)));
typedef int          int32x8  __attribute__((ext_vector_type(8)));
typedef int          int32x4  __attribute__((ext_vector_type(4)));

#define WMMA_F16(a, b, c) \
  __builtin_amdgcn_wmma_f32_16x16x32_f16(false, (a), false, (b), (short)0, (c), false, false)

static __device__ __forceinline__ v16h pack16(v8h lo, v8h hi) {
  return __builtin_shufflevector(lo, hi, 0,1,2,3,4,5,6,7,8,9,10,11,12,13,14,15);
}

// A fragment (16x32 f16): lanes 0-15 hold M=0..15, K=0..15;
// lanes 16-31 hold K=16..31 (contiguous). Source f32 row-major.
static __device__ __forceinline__ v16h a_frag_f32(const float* base, int ld,
                                                  int mBase, int kBase, int lane) {
  const float* p = base + (size_t)(mBase + (lane & 15)) * ld + kBase + ((lane >> 4) << 4);
  v16h a;
#pragma unroll
  for (int i = 0; i < 8; ++i) a[i] = (_Float16)p[i];
#pragma unroll
  for (int i = 0; i < 8; ++i) a[8 + i] = (_Float16)p[8 + i];
  return a;
}

// Same A fragment from f16 row-major source (LDS or global): two b128 loads.
static __device__ __forceinline__ v16h a_frag_f16(const _Float16* base, int ld,
                                                  int mBase, int kBase, int lane) {
  const _Float16* p = base + (size_t)(mBase + (lane & 15)) * ld + kBase + ((lane >> 4) << 4);
  v8h lo = *(const v8h*)p;
  v8h hi = *(const v8h*)(p + 8);
  return pack16(lo, hi);
}

// Fragment-packed B load: lane's 16 halfs are contiguous 32B.
static __device__ __forceinline__ v16h b_frag_packed(const _Float16* frags,
                                                     int fragIdx, int lane) {
  const _Float16* p = frags + (size_t)fragIdx * 512 + lane * 16;
  return pack16(*(const v8h*)p, *(const v8h*)(p + 8));
}

// ------------- W swizzle: f32 [K][N] row-major -> f16 B fragments -----------
// Fragment f = kt*(N/16)+nt holds 32x16 tile; lane n=lane&15,
// lanes 0-15 K=kt*32+0..15, lanes 16-31 K=+16..31, contiguous per lane.
__global__ __launch_bounds__(256)
void w_swizzle(const float* __restrict__ W, _Float16* __restrict__ out,
               int K, int N) {
  const int frag = blockIdx.x * 8 + (threadIdx.x >> 5);
  const int lane = threadIdx.x & 31;
  const int ntN  = N >> 4;
  const int kt   = frag / ntN;
  const int nt   = frag % ntN;
  const int n    = nt * 16 + (lane & 15);
  const int k0   = kt * 32 + ((lane >> 4) << 4);
  _Float16* dst  = out + (size_t)frag * 512 + lane * 16;
#pragma unroll
  for (int i = 0; i < 16; ++i) dst[i] = (_Float16)W[(size_t)(k0 + i) * N + n];
}

// ---------------- Kernel A: Xh = X @ W_xh + b_h  ([32768,512]x[512,1024]) ---
// 2x4 register blocking: wave owns 32 rows x 64 cols (8 accumulators).
__global__ __launch_bounds__(256)
void rnn_xh_gemm(const float* __restrict__ X, const _Float16* __restrict__ WxhF,
                 const float* __restrict__ bh, _Float16* __restrict__ Xh) {
  const int lane  = threadIdx.x & 31;
  const int wave  = threadIdx.x >> 5;
  const int mBase = blockIdx.x * 32;
  const int ng    = blockIdx.y * 8 + wave;   // 16 col-groups of 64
  const int nt0   = ng * 4;                  // first of 4 n-tiles
  v8f acc0[4] = {}, acc1[4] = {};
#pragma unroll 2
  for (int kt = 0; kt < 16; ++kt) {          // K = 512
    v16h a0 = a_frag_f32(X, 512, mBase,      kt * 32, lane);
    v16h a1 = a_frag_f32(X, 512, mBase + 16, kt * 32, lane);
#pragma unroll
    for (int j = 0; j < 4; ++j) {
      v16h b = b_frag_packed(WxhF, kt * 64 + nt0 + j, lane);
      acc0[j] = WMMA_F16(a0, b, acc0[j]);
      acc1[j] = WMMA_F16(a1, b, acc1[j]);
    }
  }
#pragma unroll
  for (int j = 0; j < 4; ++j) {
    const int   n    = (nt0 + j) * 16 + (lane & 15);
    const float bias = bh[n];
#pragma unroll
    for (int r = 0; r < 8; ++r) {            // C: VGPR r -> M = r (+8 hi half)
      const int m = mBase + r + ((lane >> 4) << 3);
      Xh[(size_t)m * 1024 + n]        = (_Float16)(acc0[j][r] + bias);
      Xh[(size_t)(m + 16) * 1024 + n] = (_Float16)(acc1[j][r] + bias);
    }
  }
}

// ---------------- Kernel B: persistent recurrence over T=512 steps ----------
// 16 WGs x 256 threads. WG g owns hidden cols [g*64, g*64+64).
// LDS: 128KB W_hh B-fragments (offset 0) + 128KB h_{t-1} stage (offset 131072).
// h staging done by the Tensor Data Mover (one tensor_load_to_lds per step).
__global__ __launch_bounds__(256)
void rnn_recurrence(const _Float16* __restrict__ Xh, const float* __restrict__ Whh,
                    _Float16* __restrict__ Hall, unsigned* __restrict__ counter) {
  extern __shared__ char smem[];
  _Float16* wfrag = (_Float16*)smem;              // 128 frags x (32 lanes x 16 halfs)
  _Float16* hbuf  = (_Float16*)(smem + 131072);   // [64][1024] f16
  const int tid  = threadIdx.x;
  const int lane = tid & 31;
  const int wave = tid >> 5;
  const int wg   = blockIdx.x;                    // 0..15

  // One-time: swizzle W_hh[:, wg*64 .. +64) (f32) into fragment-ready f16 LDS.
  for (int slot = tid; slot < 128 * 32; slot += 256) {
    const int frag  = slot >> 5;                  // frag = kt*4 + ntL
    const int flane = slot & 31;
    const int kt    = frag >> 2;
    const int ntL   = frag & 3;
    const int n     = wg * 64 + ntL * 16 + (flane & 15);
    const int k0    = kt * 32 + ((flane >> 4) << 4);
    _Float16* dst   = wfrag + (size_t)frag * 512 + flane * 16;
#pragma unroll
    for (int i = 0; i < 16; ++i) dst[i] = (_Float16)Whh[(size_t)(k0 + i) * 1024 + n];
  }
  __syncthreads();

  const int mt   = wave & 3;                      // batch-row tile (M=64 -> 4 tiles)
  const int ntL0 = (wave >> 2) * 2;               // this wave's two local n-tiles
  const int ntL1 = ntL0 + 1;
  const int nc0  = wg * 64 + ntL0 * 16 + (lane & 15);
  const int nc1  = nc0 + 16;

  for (int t = 0; t < 512; ++t) {
    // Prefetch this step's Xh lines (consumed in the epilogue).
#pragma unroll
    for (int r = 0; r < 8; ++r) {
      const int b = mt * 16 + r + ((lane >> 4) << 3);
      __builtin_prefetch(&Xh[((size_t)b * 512 + t) * 1024 + nc0], 0, 0);
    }
    v8f acc0 = {}, acc1 = {};
    if (t > 0) {
      if (wave == 0) {
        // TDM: DMA h_{t-1} ([64][1024] f16, 128KB) global -> LDS hbuf.
        const unsigned long long ga =
            (unsigned long long)(const void*)(Hall + (size_t)(t - 1) * 65536);
        uint32x4 g0;
        g0[0] = 1u;                                   // count=1, user descriptor
        g0[1] = 131072u;                              // lds_addr (hbuf byte offset)
        g0[2] = (unsigned)(ga & 0xFFFFFFFFu);         // global_addr[31:0]
        g0[3] = (unsigned)((ga >> 32) & 0x01FFFFFFu)  // global_addr[56:32]
              | (2u << 30);                           // type = 2 ("image")
        int32x8 g1;
        g1[0] = (int)(1u << 16);      // workgroup_mask=0 | data_size=1 (2B)
        g1[1] = (int)(1024u << 16);   // tensor_dim0[15:0]=1024 at bits 63:48
        g1[2] = (int)(64u << 16);     // tensor_dim0 hi=0 | tensor_dim1=64
        g1[3] = (int)(1024u << 16);   // tensor_dim1 hi=0 | tile_dim0=1024
        g1[4] = 64;                   // tile_dim1=64 | tile_dim2=0
        g1[5] = 1024;                 // tensor_dim0_stride lo = 1024 elements
        g1[6] = 0;
        g1[7] = 0;
        int32x4 gz4 = {0, 0, 0, 0};   // groups 2/3 unused (2D tensor)
        int32x8 gz8 = {0, 0, 0, 0, 0, 0, 0, 0};
        __builtin_amdgcn_tensor_load_to_lds(g0, g1, gz4, gz4, gz8, 0);
        __builtin_amdgcn_s_wait_tensorcnt(0);
      }
      __syncthreads();                // hbuf now holds h_{t-1}
#pragma unroll 4
      for (int kt = 0; kt < 32; ++kt) {             // K = 1024
        v16h a  = a_frag_f16(hbuf, 1024, mt * 16, kt * 32, lane);
        v16h b0 = b_frag_packed(wfrag, kt * 4 + ntL0, lane);
        v16h b1 = b_frag_packed(wfrag, kt * 4 + ntL1, lane);
        acc0 = WMMA_F16(a, b0, acc0);               // independent accs -> co-issue
        acc1 = WMMA_F16(a, b1, acc1);
      }
    }
    // Epilogue: h_t = tanh(h_{t-1}@Whh + Xh[:,t]); store f16 slice.
    _Float16* hout = Hall + (size_t)t * 65536;
#pragma unroll
    for (int r = 0; r < 8; ++r) {
      const int    b    = mt * 16 + r + ((lane >> 4) << 3); // batch index
      const size_t xrow = ((size_t)b * 512 + t) * 1024;     // Xh is [B][T][H]
      const float  h0   = tanhf(acc0[r] + (float)Xh[xrow + nc0]);
      const float  h1   = tanhf(acc1[r] + (float)Xh[xrow + nc1]);
      hout[(size_t)b * 1024 + nc0] = (_Float16)h0;
      hout[(size_t)b * 1024 + nc1] = (_Float16)h1;
    }
    // Device-scope barrier across the 16 WGs (monotonic counter, fenced).
    if (t < 511) {
      __syncthreads();
      if (tid == 0) {
        __builtin_amdgcn_fence(__ATOMIC_RELEASE, "agent");
        atomicAdd(counter, 1u);
        const unsigned target = 16u * (unsigned)(t + 1);
        while (__hip_atomic_load(counter, __ATOMIC_RELAXED, __HIP_MEMORY_SCOPE_AGENT) < target)
          __builtin_amdgcn_s_sleep(2);
        __builtin_amdgcn_fence(__ATOMIC_ACQUIRE, "agent");
      }
      __syncthreads();
    }
  }
}

// ---------------- Kernel C: Y = Hall @ W_hy + b_y ([32768,1024]x[1024,512]) -
__global__ __launch_bounds__(256)
void rnn_y_gemm(const _Float16* __restrict__ Hall, const _Float16* __restrict__ WhyF,
                const float* __restrict__ by, float* __restrict__ out) {
  const int lane  = threadIdx.x & 31;
  const int wave  = threadIdx.x >> 5;
  const int mBase = blockIdx.x * 32;         // row = t*64 + b (Hall is [T][B][H])
  const int nt0   = wave * 4;                // 8 waves cover all 32 n-tiles
  v8f acc0[4] = {}, acc1[4] = {};
#pragma unroll 2
  for (int kt = 0; kt < 32; ++kt) {          // K = 1024
    v16h a0 = a_frag_f16(Hall, 1024, mBase,      kt * 32, lane);
    v16h a1 = a_frag_f16(Hall, 1024, mBase + 16, kt * 32, lane);
#pragma unroll
    for (int j = 0; j < 4; ++j) {
      v16h b = b_frag_packed(WhyF, kt * 32 + nt0 + j, lane);
      acc0[j] = WMMA_F16(a0, b, acc0[j]);
      acc1[j] = WMMA_F16(a1, b, acc1[j]);
    }
  }
#pragma unroll
  for (int j = 0; j < 4; ++j) {
    const int   n    = (nt0 + j) * 16 + (lane & 15);
    const float bias = by[n];
#pragma unroll
    for (int r = 0; r < 8; ++r) {
      const int m0 = mBase + r + ((lane >> 4) << 3);
      const int t0 = m0 >> 6, b0 = m0 & 63;
      const int m1 = m0 + 16;
      const int t1 = m1 >> 6, b1 = m1 & 63;
      out[((size_t)b0 * 512 + t0) * 512 + n] = acc0[j][r] + bias;  // Y is [B][T][Y]
      out[((size_t)b1 * 512 + t1) * 512 + n] = acc1[j][r] + bias;
    }
  }
}

extern "C" void kernel_launch(void* const* d_in, const int* in_sizes, int n_in,
                              void* d_out, int out_size, void* d_ws, size_t ws_size,
                              hipStream_t stream) {
  (void)in_sizes; (void)n_in; (void)out_size; (void)ws_size;
  const float* X    = (const float*)d_in[0];
  const float* W_hh = (const float*)d_in[1];
  const float* W_xh = (const float*)d_in[2];
  const float* b_h  = (const float*)d_in[3];
  const float* W_hy = (const float*)d_in[4];
  const float* b_y  = (const float*)d_in[5];
  float* out = (float*)d_out;

  // Workspace: [counter 256B][Xh f16 64MB][Hall f16 64MB][WxhF 1MB][WhyF 1MB]
  char*     ws      = (char*)d_ws;
  unsigned* counter = (unsigned*)ws;
  _Float16* Xh      = (_Float16*)(ws + 256);
  _Float16* Hall    = (_Float16*)(ws + 256 + (size_t)67108864);
  _Float16* WxhF    = (_Float16*)(ws + 256 + (size_t)67108864 * 2);
  _Float16* WhyF    = (_Float16*)(ws + 256 + (size_t)67108864 * 2 + 1048576);

  (void)hipMemsetAsync(counter, 0, 256, stream);  // graph-capture-safe memset node

  w_swizzle<<<dim3(128), 256, 0, stream>>>(W_xh, WxhF, 512, 1024);  // 1024 frags
  w_swizzle<<<dim3(128), 256, 0, stream>>>(W_hy, WhyF, 1024, 512);  // 1024 frags

  rnn_xh_gemm   <<<dim3(1024, 2), 256, 0,      stream>>>(X, WxhF, b_h, Xh);
  rnn_recurrence<<<dim3(16),      256, 262144, stream>>>(Xh, W_hh, Hall, counter);
  rnn_y_gemm    <<<dim3(1024),    256, 0,      stream>>>(Hall, WhyF, b_y, out);
}
